// Predictor_17824114278932
// MI455X (gfx1250) — compile-verified, matrix operand order checked by reading, not run on hardware
//
#include <hip/hip_runtime.h>
#include <hip/hip_bf16.h>
#include <math.h>

#define B_ 16
#define T_ 2048
#define H_ 512
#define K_ 15
#define LN_EPS 1e-5f

// GEMM tiling
#define MB 128
#define NB 64
#define KC 32
#define LDT 40   // padded LDS stride in halves (80B, 16B-aligned, bank-spread)

typedef __attribute__((ext_vector_type(16))) _Float16 v16h;
typedef __attribute__((ext_vector_type(8)))  _Float16 v8h;
typedef __attribute__((ext_vector_type(8)))  float    v8f;
typedef __attribute__((ext_vector_type(4)))  unsigned int u32x4;
typedef __attribute__((ext_vector_type(8)))  int          i32x8;
typedef __attribute__((ext_vector_type(4)))  int          i32x4;

#if defined(__has_builtin)
#if __has_builtin(__builtin_amdgcn_tensor_load_to_lds) && \
    __has_builtin(__builtin_amdgcn_s_wait_tensorcnt)
#define USE_TDM 1
#endif
#endif
#ifndef USE_TDM
#define USE_TDM 0
#endif

static __device__ __forceinline__ float sigmoidf_(float v) {
    return 1.0f / (1.0f + __expf(-v));
}

// ---------------------------------------------------------------------------
// Zero the whole output buffer (non-fired rows must be zero; token sums too).
// ---------------------------------------------------------------------------
__global__ void zero_out_kernel(float* __restrict__ p, long n) {
    long i = (long)blockIdx.x * blockDim.x + threadIdx.x;
    long stride = (long)gridDim.x * blockDim.x;
    for (; i < n; i += stride) p[i] = 0.0f;
}

// ---------------------------------------------------------------------------
// Collapse pw2 + lin into one H-vector:  weff[h] = sum_o lin_w[o]*pw2_w[o,h]
// ---------------------------------------------------------------------------
__global__ void prep_weff_kernel(const float* __restrict__ pw2_w,
                                 const float* __restrict__ pw2_b,
                                 const float* __restrict__ lin_w,
                                 const float* __restrict__ lin_b,
                                 float* __restrict__ weff,
                                 float* __restrict__ beff) {
    int h = threadIdx.x;  // blockDim = 512
    float acc = 0.0f;
    for (int o = 0; o < H_; o++) acc += pw2_w[(size_t)o * H_ + h] * lin_w[o];
    weff[h] = acc;
    if (h == 0) {
        float bb = lin_b[0];
        for (int o = 0; o < H_; o++) bb += pw2_b[o] * lin_w[o];
        beff[0] = bb;
    }
}

// f32 -> f16 bulk converts (pw1_w and x)
__global__ void convert_f16_kernel(const float* __restrict__ src,
                                   _Float16* __restrict__ dst) {
    size_t i = (size_t)blockIdx.x * 256 + threadIdx.x;
    dst[i] = (_Float16)src[i];
}

// ---------------------------------------------------------------------------
// TDM 2D tile load: f16 tile (tile_k x tile_rows), row stride 512 elements,
// into LDS with hardware padding of 4 DWORDs after every 16 DWORDs
// (=> LDS row stride of 40 halves, matching fragment reads).
// ---------------------------------------------------------------------------
#if USE_TDM
static __device__ __forceinline__ void tdm_load_tile(
    unsigned lds_addr, const _Float16* gptr, int tile_rows, int tensor_rows) {
    unsigned long long ga = (unsigned long long)(uintptr_t)gptr;
    u32x4 g0;
    g0[0] = 1u;  // count=1, is_restore=0, gather_mode=0
    g0[1] = lds_addr;
    g0[2] = (unsigned)(ga & 0xFFFFFFFFull);
    g0[3] = (unsigned)((ga >> 32) & 0x1FFFFFFull) | (2u << 30);  // type=2
    i32x8 g1;
    g1[0] = (1 << 16)    // data_size = 2 bytes
          | (1 << 20)    // pad_enable
          | (3 << 22)    // pad_interval: 16 DWORDs
          | (3 << 25);   // pad_amount: 4 DWORDs
    g1[1] = (H_ & 0xFFFF) << 16;                       // tensor_dim0 lo16
    g1[2] = ((H_ >> 16) & 0xFFFF) |
            ((tensor_rows & 0xFFFF) << 16);            // dim0 hi | dim1 lo
    g1[3] = ((tensor_rows >> 16) & 0xFFFF) |
            ((KC & 0xFFFF) << 16);                     // dim1 hi | tile_dim0
    g1[4] = (tile_rows & 0xFFFF);                      // tile_dim1, tile_dim2=0
    g1[5] = H_;                                        // dim0_stride lo32
    g1[6] = 0;
    g1[7] = 0;
    i32x4 z4 = {0, 0, 0, 0};
    i32x8 z8 = {0, 0, 0, 0, 0, 0, 0, 0};
    __builtin_amdgcn_tensor_load_to_lds(g0, g1, z4, z4, z8, 0);
}
static __device__ __forceinline__ unsigned lds_off_(const void* p) {
    return (unsigned)(unsigned long long)(uintptr_t)p;
}
#endif

// ---------------------------------------------------------------------------
// GEMM + GLU:  h_glu[m, n] = (X@W1^T + b)[m,n] * sigmoid((X@W1^T + b)[m,n+512])
// M = B*T, K = 512.  Block: 256 threads (8 waves), tile 128 rows x 64 cols,
// both GLU halves.  TDM double-buffered LDS staging + WMMA f32_16x16x32_f16.
// ---------------------------------------------------------------------------
__global__ __launch_bounds__(256) void gemm_glu_kernel(
    const _Float16* __restrict__ xh, const _Float16* __restrict__ w1h,
    const float* __restrict__ pw1_b, float* __restrict__ h_glu) {
    __shared__ _Float16 As[2][MB * LDT];
    __shared__ _Float16 Bs0[2][NB * LDT];
    __shared__ _Float16 Bs1[2][NB * LDT];

    const int tid = threadIdx.x;
    const int m0 = blockIdx.x * MB;
    const int n0 = blockIdx.y * NB;
    const int wv = tid >> 5;
    const int lane = tid & 31;
    const int mi = wv & 3;   // 4 M-subtiles of 32
    const int ni = wv >> 2;  // 2 N-subtiles of 32
    const int rl = lane & 15;
    const int kb = (lane < 16) ? 0 : 8;  // wave32 16-bit fragment K packing

    v8f z = {0.f, 0.f, 0.f, 0.f, 0.f, 0.f, 0.f, 0.f};
    v8f cA[2][2], cB[2][2];
#pragma unroll
    for (int i = 0; i < 2; i++)
#pragma unroll
        for (int j = 0; j < 2; j++) { cA[i][j] = z; cB[i][j] = z; }

    int buf = 0;

#if USE_TDM
    if (wv == 0) {
        tdm_load_tile(lds_off_(&As[0][0]), xh + (size_t)m0 * H_, MB, B_ * T_);
        tdm_load_tile(lds_off_(&Bs0[0][0]), w1h + (size_t)n0 * H_, NB, 2 * H_);
        tdm_load_tile(lds_off_(&Bs1[0][0]), w1h + (size_t)(n0 + H_) * H_, NB,
                      2 * H_);
        __builtin_amdgcn_s_wait_tensorcnt(0);
    }
    __syncthreads();
#else
    const int arow = tid >> 1;        // 0..127
    const int akh  = (tid & 1) * 16;  // 0 or 16
    const int bn   = tid >> 2;        // 0..63
    const int bks  = (tid & 3) * 8;   // 0,8,16,24
#endif

    for (int k0 = 0; k0 < H_; k0 += KC) {
#if USE_TDM
        // Prefetch next K-slab into the other buffer while we compute.
        if (wv == 0 && (k0 + KC) < H_) {
            int kn = k0 + KC;
            tdm_load_tile(lds_off_(&As[buf ^ 1][0]),
                          xh + (size_t)m0 * H_ + kn, MB, B_ * T_);
            tdm_load_tile(lds_off_(&Bs0[buf ^ 1][0]),
                          w1h + (size_t)n0 * H_ + kn, NB, 2 * H_);
            tdm_load_tile(lds_off_(&Bs1[buf ^ 1][0]),
                          w1h + (size_t)(n0 + H_) * H_ + kn, NB, 2 * H_);
        }
#else
        {  // cooperative synchronous staging (fallback)
            v8h a0 = *(const v8h*)(xh + (size_t)(m0 + arow) * H_ + k0 + akh);
            v8h a1 = *(const v8h*)(xh + (size_t)(m0 + arow) * H_ + k0 + akh + 8);
            *(v8h*)&As[0][arow * LDT + akh] = a0;
            *(v8h*)&As[0][arow * LDT + akh + 8] = a1;
            v8h vb0 = *(const v8h*)(w1h + (size_t)(n0 + bn) * H_ + k0 + bks);
            v8h vb1 = *(const v8h*)(w1h + (size_t)(n0 + H_ + bn) * H_ + k0 + bks);
            *(v8h*)&Bs0[0][bn * LDT + bks] = vb0;
            *(v8h*)&Bs1[0][bn * LDT + bks] = vb1;
        }
        __syncthreads();
#endif

        const _Float16* Ab = &As[buf][0];
        const _Float16* B0b = &Bs0[buf][0];
        const _Float16* B1b = &Bs1[buf][0];

        v16h a[2], b0[2], b1[2];
#pragma unroll
        for (int i = 0; i < 2; i++) {
            int r = mi * 32 + i * 16 + rl;
            v8h lo = *(const v8h*)&Ab[r * LDT + kb];
            v8h hi = *(const v8h*)&Ab[r * LDT + kb + 16];
            a[i] = __builtin_shufflevector(lo, hi, 0, 1, 2, 3, 4, 5, 6, 7, 8, 9,
                                           10, 11, 12, 13, 14, 15);
        }
#pragma unroll
        for (int j = 0; j < 2; j++) {
            int c = ni * 32 + j * 16 + rl;
            v8h lo0 = *(const v8h*)&B0b[c * LDT + kb];
            v8h hi0 = *(const v8h*)&B0b[c * LDT + kb + 16];
            b0[j] = __builtin_shufflevector(lo0, hi0, 0, 1, 2, 3, 4, 5, 6, 7, 8,
                                            9, 10, 11, 12, 13, 14, 15);
            v8h lo1 = *(const v8h*)&B1b[c * LDT + kb];
            v8h hi1 = *(const v8h*)&B1b[c * LDT + kb + 16];
            b1[j] = __builtin_shufflevector(lo1, hi1, 0, 1, 2, 3, 4, 5, 6, 7, 8,
                                            9, 10, 11, 12, 13, 14, 15);
        }
#pragma unroll
        for (int i = 0; i < 2; i++)
#pragma unroll
            for (int j = 0; j < 2; j++) {
                cA[i][j] = __builtin_amdgcn_wmma_f32_16x16x32_f16(
                    false, a[i], false, b0[j], (short)0, cA[i][j], false, false);
                cB[i][j] = __builtin_amdgcn_wmma_f32_16x16x32_f16(
                    false, a[i], false, b1[j], (short)0, cB[i][j], false, false);
            }

#if USE_TDM
        if (wv == 0) __builtin_amdgcn_s_wait_tensorcnt(0);
        __syncthreads();
        buf ^= 1;
#else
        __syncthreads();
#endif
    }

    // Epilogue: bias + GLU, write f32
#pragma unroll
    for (int j = 0; j < 2; j++) {
        int ng = n0 + ni * 32 + j * 16 + rl;
        float ba = pw1_b[ng];
        float bb = pw1_b[ng + 512];
#pragma unroll
        for (int i = 0; i < 2; i++) {
            int mbase = m0 + mi * 32 + i * 16 + ((lane >> 4) << 3);
#pragma unroll
            for (int r = 0; r < 8; r++) {
                float va = cA[i][j][r] + ba;
                float vb = cB[i][j][r] + bb;
                h_glu[(size_t)(mbase + r) * H_ + ng] = va * sigmoidf_(vb);
            }
        }
    }
}

// ---------------------------------------------------------------------------
// Block-wide sum over 256 threads (LDS tree). All threads get the result.
// ---------------------------------------------------------------------------
static __device__ __forceinline__ float block_reduce_sum(float v, float* red) {
    int tid = threadIdx.x;
    red[tid] = v;
    __syncthreads();
#pragma unroll
    for (int off = 128; off > 0; off >>= 1) {
        if (tid < off) red[tid] += red[tid + off];
        __syncthreads();
    }
    float r = red[0];
    __syncthreads();
    return r;
}

// ---------------------------------------------------------------------------
// Depthwise causal conv (K=15) + LayerNorm + swish + dot(weff) -> alphas.
// One block of 256 threads per (b,t); h_glu lives in L2 (67MB < 192MB).
// ---------------------------------------------------------------------------
__global__ __launch_bounds__(256) void conv_ln_alpha_kernel(
    const float* __restrict__ h_glu, const float* __restrict__ dw_w,
    const float* __restrict__ dw_b, const float* __restrict__ ln_g,
    const float* __restrict__ ln_b, const float* __restrict__ weff,
    const float* __restrict__ beff, const int* __restrict__ x_lens,
    float* __restrict__ alphas) {
    __shared__ float red[256];
    const int b = blockIdx.x >> 11;
    const int t = blockIdx.x & (T_ - 1);
    const int tid = threadIdx.x;
    const float* base = h_glu + (size_t)b * T_ * H_;

    float v[2];
#pragma unroll
    for (int e2 = 0; e2 < 2; e2++) {
        int h = tid + e2 * 256;
        float acc = dw_b[h];
#pragma unroll
        for (int k = 0; k < K_; k++) {
            int ts = t - (K_ - 1) + k;
            if (ts >= 0) acc += base[(size_t)ts * H_ + h] * dw_w[h * K_ + k];
        }
        v[e2] = acc;
    }
    float mean = block_reduce_sum(v[0] + v[1], red) * (1.0f / H_);
    float d0 = v[0] - mean, d1 = v[1] - mean;
    float var = block_reduce_sum(d0 * d0 + d1 * d1, red) * (1.0f / H_);
    float rs = rsqrtf(var + LN_EPS);

    float p = 0.0f;
#pragma unroll
    for (int e2 = 0; e2 < 2; e2++) {
        int h = tid + e2 * 256;
        float nv = (v[e2] - mean) * rs * ln_g[h] + ln_b[h];
        float sw = nv * sigmoidf_(nv);
        p += sw * weff[h];
    }
    float logit = block_reduce_sum(p, red);
    if (tid == 0) {
        float a = sigmoidf_(logit + beff[0]);
        if (t >= x_lens[b]) a = 0.0f;
        alphas[(size_t)b * T_ + t] = a;
    }
}

// ---------------------------------------------------------------------------
// CIF scalar scan, one thread per batch. Produces per-t contribution weight
// w_t (dist at fires, alpha otherwise), leftover e_t (remain at fires), fire
// positions, fire counts, and token sums (written straight into d_out).
// ---------------------------------------------------------------------------
__global__ void cif_scan_kernel(const float* __restrict__ alphas,
                                float* __restrict__ wbuf,
                                float* __restrict__ ebuf,
                                int* __restrict__ fpos,
                                int* __restrict__ nfires,
                                float* __restrict__ tok) {
    int b = threadIdx.x;
    if (b >= B_) return;
    const float* ab = alphas + (size_t)b * T_;
    float* wb = wbuf + (size_t)b * T_;
    float* eb = ebuf + (size_t)b * T_;
    int* fp = fpos + (size_t)b * T_;
    float integ = 0.0f, sum = 0.0f;
    int j = 0;
    for (int t4 = 0; t4 < T_ / 4; t4++) {
        float4 av = ((const float4*)ab)[t4];
        float aq[4] = {av.x, av.y, av.z, av.w};
        float wq[4], eq[4];
#pragma unroll
        for (int q = 0; q < 4; q++) {
            float a = aq[q];
            sum += a;
            float dist = 1.0f - integ;
            integ += a;
            bool fire = (integ >= 1.0f);
            float cur = fire ? dist : a;
            wq[q] = cur;
            eq[q] = fire ? (a - cur) : 0.0f;
            if (fire) { fp[j++] = t4 * 4 + q; integ -= 1.0f; }
        }
        ((float4*)wb)[t4] = make_float4(wq[0], wq[1], wq[2], wq[3]);
        ((float4*)eb)[t4] = make_float4(eq[0], eq[1], eq[2], eq[3]);
    }
    nfires[b] = j;
    tok[b] = sum;
}

// ---------------------------------------------------------------------------
// Deterministic gather: fired frame j = e[prev_fire]*x[prev_fire]
//   + sum_{prev<t<=t_j} w_t*x[t].  Writes forward + H-flipped outputs.
// Rows j >= nfires stay zero (pre-zeroed). One block per (b, j).
// ---------------------------------------------------------------------------
__global__ __launch_bounds__(256) void gather_kernel(
    const float* __restrict__ x, const float* __restrict__ wbuf,
    const float* __restrict__ ebuf, const int* __restrict__ fpos,
    const int* __restrict__ nfires, float* __restrict__ out) {
    const int b = blockIdx.x >> 11;
    const int j = blockIdx.x & (T_ - 1);
    if (j >= nfires[b]) return;
    const int tid = threadIdx.x;
    const int tj = fpos[(size_t)b * T_ + j];
    const int tp = (j > 0) ? fpos[(size_t)b * T_ + j - 1] : -1;
    const float* xb = x + (size_t)b * T_ * H_;
    const float* wb = wbuf + (size_t)b * T_;
    const int h0 = tid, h1 = tid + 256;
    float acc0 = 0.0f, acc1 = 0.0f;
    if (j > 0) {
        float ew = ebuf[(size_t)b * T_ + tp];
        acc0 = ew * xb[(size_t)tp * H_ + h0];
        acc1 = ew * xb[(size_t)tp * H_ + h1];
    }
    for (int t = tp + 1; t <= tj; t++) {
        float wt = wb[t];
        acc0 += wt * xb[(size_t)t * H_ + h0];
        acc1 += wt * xb[(size_t)t * H_ + h1];
    }
    const size_t o = (size_t)b * T_ * H_ + (size_t)j * H_;
    out[o + h0] = acc0;
    out[o + h1] = acc1;
    float* rout = out + (size_t)B_ * T_ * H_ + B_;  // r_fired_frames region
    rout[o + (H_ - 1 - h0)] = acc0;
    rout[o + (H_ - 1 - h1)] = acc1;
}

// ---------------------------------------------------------------------------
extern "C" void kernel_launch(void* const* d_in, const int* in_sizes, int n_in,
                              void* d_out, int out_size, void* d_ws,
                              size_t ws_size, hipStream_t stream) {
    const float* x     = (const float*)d_in[0];
    const int*   xlens = (const int*)d_in[1];
    const float* pw1_w = (const float*)d_in[2];
    const float* pw1_b = (const float*)d_in[3];
    const float* dw_w  = (const float*)d_in[4];
    const float* dw_b  = (const float*)d_in[5];
    const float* ln_g  = (const float*)d_in[6];
    const float* ln_b  = (const float*)d_in[7];
    const float* pw2_w = (const float*)d_in[8];
    const float* pw2_b = (const float*)d_in[9];
    const float* lin_w = (const float*)d_in[10];
    const float* lin_b = (const float*)d_in[11];
    float* out = (float*)d_out;

    char* ws = (char*)d_ws;
    size_t off = 0;
    auto take = [&](size_t bytes) -> char* {
        char* p = ws + off;
        off = (off + bytes + 255) & ~(size_t)255;
        return p;
    };
    _Float16* w1h   = (_Float16*)take((size_t)2 * H_ * H_ * sizeof(_Float16));
    _Float16* xh    = (_Float16*)take((size_t)B_ * T_ * H_ * sizeof(_Float16));
    float*    weff  = (float*)take(H_ * sizeof(float));
    float*    beff  = (float*)take(16);
    float*    h_glu = (float*)take((size_t)B_ * T_ * H_ * sizeof(float));
    float*    alphas= (float*)take((size_t)B_ * T_ * sizeof(float));
    float*    wbuf  = (float*)take((size_t)B_ * T_ * sizeof(float));
    float*    ebuf  = (float*)take((size_t)B_ * T_ * sizeof(float));
    int*      fpos  = (int*)take((size_t)B_ * T_ * sizeof(int));
    int*      nfires= (int*)take(B_ * sizeof(int));

    zero_out_kernel<<<4096, 256, 0, stream>>>(out, (long)out_size);
    prep_weff_kernel<<<1, 512, 0, stream>>>(pw2_w, pw2_b, lin_w, lin_b, weff,
                                            beff);
    convert_f16_kernel<<<(2 * H_ * H_) / 256, 256, 0, stream>>>(pw1_w, w1h);
    convert_f16_kernel<<<(B_ * T_ * H_) / 256, 256, 0, stream>>>(x, xh);
    gemm_glu_kernel<<<dim3((B_ * T_) / MB, H_ / NB), 256, 0, stream>>>(
        xh, w1h, pw1_b, h_glu);
    conv_ln_alpha_kernel<<<B_ * T_, 256, 0, stream>>>(
        h_glu, dw_w, dw_b, ln_g, ln_b, weff, beff, xlens, alphas);
    cif_scan_kernel<<<1, 32, 0, stream>>>(alphas, wbuf, ebuf, fpos, nfires,
                                          out + (size_t)B_ * T_ * H_);
    gather_kernel<<<B_ * T_, 256, 0, stream>>>(x, wbuf, ebuf, fpos, nfires, out);
}
